// RNNDecoder_82411832475670
// MI455X (gfx1250) — compile-verified, matrix operand order checked by reading, not run on hardware
//
#include <hip/hip_runtime.h>
#include <hip/hip_bf16.h>

// ---------------------------------------------------------------------------
// MI455X (gfx1250) RNN decoder with memory networks.
//  * All dense matmuls -> bf16 WMMA (v_wmma_f32_16x16x32_bf16), fp32 accum.
//  * K@Wk recomputation replaced by exact rank-1 incremental updates
//    (K_new = K + p (x) o  =>  KW_new = KW + p (x) (o@Wk)); ~60MB of state
//    stays L2-resident (192MB L2) across the 64 sequential steps.
//  * kbt memory is static: its K@Wk+b is hoisted out of the T-loop.
//  * GEMM: double-buffered LDS tiles; all 8 global b128 loads of the next
//    K-tile are issued before the WMMA sequence (latency hidden behind MMA);
//    B tile stored transposed so every WMMA fragment is 2x ds_load_b128.
// ---------------------------------------------------------------------------

typedef __attribute__((ext_vector_type(16))) __bf16 v16bf;
typedef __attribute__((ext_vector_type(8)))  float  v8f;

#define Bb   32
#define Tt   64
#define Hd   512
#define Ed   512
#define Vv   32000
#define Ll   128      // LD == LK == LT
#define OUTIN 2560
#define RNNIN 2048

#define TILE_M 64
#define TILE_N 64
#define TILE_K 32
#define LDSP   40     // LDS row pitch in bf16 (80B: keeps 16B-aligned slices)

__device__ __forceinline__ float sigmoidf_(float x) { return 1.0f / (1.0f + __expf(-x)); }

// ---------------------------------------------------------------------------
// Generic WMMA GEMM: C[cbase + m*ldc + n] = sum_k A[abase+m*lda+k] * B(k,n) + bias[n]
// B(k,n) = Bw[k*ldb + n] (transB==0) or Bw[n*ldb + k] (transB==1)
// N, K multiples of 64/32 (true for all call sites); M guarded.
// 128 threads = 4 waves; 64x64 tile; each wave does 2x2 v_wmma_f32_16x16x32_bf16.
// ---------------------------------------------------------------------------
__global__ __launch_bounds__(128)
void gemm_wmma_bf16(const float* __restrict__ A, long abase, int lda,
                    const float* __restrict__ Bw, int ldb, int transB,
                    const float* __restrict__ bias,
                    float* __restrict__ C, long cbase, int ldc,
                    int M, int N, int K)
{
    __shared__ __bf16 As[2][TILE_M][LDSP];    // [buf][m][k]
    __shared__ __bf16 BsT[2][TILE_N][LDSP];   // [buf][n][k]  (transposed in LDS)

    const int tid  = threadIdx.x;
    const int lane = tid & 31;
    const int wave = tid >> 5;
    const int bm = blockIdx.y * TILE_M;
    const int bn = blockIdx.x * TILE_N;
    const int wm = (wave >> 1) * 32;
    const int wn = (wave & 1) * 32;
    const int half = lane >> 4;
    const int idx  = lane & 15;
    const bool fullM = (bm + TILE_M) <= M;

    union Pk4  { __bf16 e[4]; uint2 u; };
    union Frag { uint4 q[2]; v16bf v; };

    float4 ra[4], rb[4];   // in-flight staging registers for one K-tile

    auto load_chunks = [&](int k0) {
        if (fullM) {
            #pragma unroll
            for (int ii = 0; ii < 4; ++ii) {
                int i = tid + ii * 128;                 // 512 float4-chunks of A
                int r = i >> 3, c4 = (i & 7) * 4;
                ra[ii] = *(const float4*)(A + abase + (long)(bm + r) * lda + (k0 + c4));
            }
        } else {
            #pragma unroll
            for (int ii = 0; ii < 4; ++ii) {
                int i = tid + ii * 128;
                int r = i >> 3, c4 = (i & 7) * 4;
                int gr = bm + r;
                float4 v4 = make_float4(0.f, 0.f, 0.f, 0.f);
                if (gr < M)
                    v4 = *(const float4*)(A + abase + (long)gr * lda + (k0 + c4));
                ra[ii] = v4;
            }
        }
        if (!transB) {
            #pragma unroll
            for (int ii = 0; ii < 4; ++ii) {
                int i = tid + ii * 128;                 // k rows (32) x 16 n-chunks
                int r = i >> 4, c4 = (i & 15) * 4;
                rb[ii] = *(const float4*)(Bw + (long)(k0 + r) * ldb + (bn + c4));
            }
        } else {
            #pragma unroll
            for (int ii = 0; ii < 4; ++ii) {
                int i = tid + ii * 128;                 // n rows (64) x 8 k-chunks
                int n = i >> 3, c4 = (i & 7) * 4;
                rb[ii] = *(const float4*)(Bw + (long)(bn + n) * ldb + (k0 + c4));
            }
        }
    };

    auto store_chunks = [&](int buf) {
        #pragma unroll
        for (int ii = 0; ii < 4; ++ii) {
            int i = tid + ii * 128;
            int r = i >> 3, c4 = (i & 7) * 4;
            Pk4 p;
            p.e[0] = (__bf16)ra[ii].x; p.e[1] = (__bf16)ra[ii].y;
            p.e[2] = (__bf16)ra[ii].z; p.e[3] = (__bf16)ra[ii].w;
            *(uint2*)&As[buf][r][c4] = p.u;
        }
        if (!transB) {
            #pragma unroll
            for (int ii = 0; ii < 4; ++ii) {
                int i = tid + ii * 128;
                int r = i >> 4, c4 = (i & 15) * 4;
                BsT[buf][c4 + 0][r] = (__bf16)rb[ii].x;
                BsT[buf][c4 + 1][r] = (__bf16)rb[ii].y;
                BsT[buf][c4 + 2][r] = (__bf16)rb[ii].z;
                BsT[buf][c4 + 3][r] = (__bf16)rb[ii].w;
            }
        } else {
            #pragma unroll
            for (int ii = 0; ii < 4; ++ii) {
                int i = tid + ii * 128;
                int n = i >> 3, c4 = (i & 7) * 4;
                Pk4 p;
                p.e[0] = (__bf16)rb[ii].x; p.e[1] = (__bf16)rb[ii].y;
                p.e[2] = (__bf16)rb[ii].z; p.e[3] = (__bf16)rb[ii].w;
                *(uint2*)&BsT[buf][n][c4] = p.u;
            }
        }
    };

    v8f acc[2][2] = {};
    const int nk = K / TILE_K;

    load_chunks(0);
    for (int kt = 0; kt < nk; ++kt) {
        const int buf = kt & 1;
        store_chunks(buf);           // waits on in-flight loads here
        __syncthreads();
        if (kt + 1 < nk)
            load_chunks((kt + 1) * TILE_K);   // issue next tile's loads now;
                                              // they complete during the WMMAs
        // ---- fragments: 2x ds_load_b128 each (ISA 7.12.2 wave32 layout:
        //      elems 0..7 = K[half*8 .. +7], elems 8..15 = K[16+half*8 .. +7]) ----
        Frag af[2], bfr[2];
        #pragma unroll
        for (int mi = 0; mi < 2; ++mi) {
            int row = wm + mi * 16 + idx;
            af[mi].q[0] = *(const uint4*)&As[buf][row][half * 8];
            af[mi].q[1] = *(const uint4*)&As[buf][row][16 + half * 8];
        }
        #pragma unroll
        for (int ni = 0; ni < 2; ++ni) {
            int col = wn + ni * 16 + idx;
            bfr[ni].q[0] = *(const uint4*)&BsT[buf][col][half * 8];
            bfr[ni].q[1] = *(const uint4*)&BsT[buf][col][16 + half * 8];
        }
        #pragma unroll
        for (int mi = 0; mi < 2; ++mi)
            #pragma unroll
            for (int ni = 0; ni < 2; ++ni)
                acc[mi][ni] = __builtin_amdgcn_wmma_f32_16x16x32_bf16(
                    false, af[mi].v, false, bfr[ni].v,
                    (short)0, acc[mi][ni], false, false);
        // One barrier per iteration is sufficient: the next overwrite of this
        // buffer is gated by the *next* iteration's barrier.
    }

    // ---- store C (C/D layout: VGPR r -> M = r + 8*half, N = idx) ----
    #pragma unroll
    for (int mi = 0; mi < 2; ++mi) {
        #pragma unroll
        for (int ni = 0; ni < 2; ++ni) {
            int n = bn + wn + ni * 16 + idx;
            float bb = bias ? bias[n] : 0.0f;
            #pragma unroll
            for (int r = 0; r < 8; ++r) {
                int m = bm + wm + mi * 16 + half * 8 + r;
                if (m < M) C[cbase + (long)m * ldc + n] = acc[mi][ni][r] + bb;
            }
        }
    }
}

// ---------------------------------------------------------------------------
// Embedding gather (float4): out_in[(b*T+t), 0:512] = embed[tok]
// ---------------------------------------------------------------------------
__global__ __launch_bounds__(256)
void embed_gather(const int* __restrict__ tokens, const float* __restrict__ embed,
                  float* __restrict__ out_in)
{
    long i = (long)blockIdx.x * 256 + threadIdx.x;   // over B*T*(E/4)
    long row = i >> 7;
    int h4 = (int)(i & 127) * 4;
    int tok = tokens[row];
    float4 v = *(const float4*)&embed[(long)tok * Ed + h4];
    *(float4*)&out_in[row * OUTIN + h4] = v;
}

// ---------------------------------------------------------------------------
// Attention scores: e[b,l] = mask ? sum_h tanh(qW[b,h] + KW[b,l,h]) * v[h] : -1e9
// One wave per (b,l); float4 streams (KW is L2-resident).
// ---------------------------------------------------------------------------
__global__ __launch_bounds__(256)
void attn_scores(const float* __restrict__ qW, const float* __restrict__ KW,
                 const float* __restrict__ vvec, const unsigned char* __restrict__ mask,
                 float* __restrict__ e)
{
    int pair = blockIdx.x * 8 + (threadIdx.x >> 5);
    int lane = threadIdx.x & 31;
    int b = pair / Ll;
    const float4* kw4 = (const float4*)(KW + (long)pair * Hd);
    const float4* q4  = (const float4*)(qW + (long)b * Hd);
    const float4* v4  = (const float4*)vvec;
    float s = 0.0f;
    #pragma unroll
    for (int h4 = lane; h4 < Hd / 4; h4 += 32) {
        float4 q = q4[h4], k = kw4[h4], w = v4[h4];
        s += tanhf(q.x + k.x) * w.x + tanhf(q.y + k.y) * w.y
           + tanhf(q.z + k.z) * w.z + tanhf(q.w + k.w) * w.w;
    }
    #pragma unroll
    for (int off = 16; off > 0; off >>= 1)
        s += __shfl_down(s, off, 32);
    if (lane == 0)
        e[pair] = mask[pair] ? s : -1e9f;
}

// ---------------------------------------------------------------------------
// Softmax over L=128 + context o = p @ Vm; optional q_out = q_in + o;
// optional strided o store (into out_in) and p store (into d_out).
// One block per batch row; float2 context accumulation (256 thr x 2 = H).
// ---------------------------------------------------------------------------
__global__ __launch_bounds__(256)
void softmax_ctx(const float* __restrict__ e, const float* __restrict__ Vm,
                 float* __restrict__ p_ws,
                 float* __restrict__ p_out, long p_stride,
                 const float* __restrict__ q_in, float* __restrict__ q_out,
                 float* __restrict__ o_out, long o_base, long o_stride)
{
    __shared__ float shp[Ll];
    __shared__ float red[256];
    int b = blockIdx.x, tid = threadIdx.x;

    float val = (tid < Ll) ? e[b * Ll + tid] : -1e30f;
    red[tid] = val; __syncthreads();
    for (int s = 128; s > 0; s >>= 1) {
        if (tid < s) red[tid] = fmaxf(red[tid], red[tid + s]);
        __syncthreads();
    }
    float mx = red[0]; __syncthreads();

    float pv = (tid < Ll) ? __expf(val - mx) : 0.0f;
    red[tid] = pv; __syncthreads();
    for (int s = 128; s > 0; s >>= 1) {
        if (tid < s) red[tid] += red[tid + s];
        __syncthreads();
    }
    float inv = 1.0f / red[0];
    __syncthreads();

    if (tid < Ll) {
        float p = pv * inv;
        shp[tid] = p;
        p_ws[b * Ll + tid] = p;
        if (p_out) p_out[(long)b * p_stride + tid] = p;
    }
    __syncthreads();

    const float2* vm2 = (const float2*)(Vm + (long)b * Ll * Hd);
    float2 acc = make_float2(0.f, 0.f);
    #pragma unroll 4
    for (int l = 0; l < Ll; ++l) {
        float2 mv = vm2[(long)l * (Hd / 2) + tid];
        acc.x += shp[l] * mv.x;
        acc.y += shp[l] * mv.y;
    }
    if (o_out) *(float2*)&o_out[o_base + (long)b * o_stride + tid * 2] = acc;
    if (q_out) {
        float2 qv = *(const float2*)&q_in[(long)b * Hd + tid * 2];
        qv.x += acc.x; qv.y += acc.y;
        *(float2*)&q_out[(long)b * Hd + tid * 2] = qv;
    }
}

// ---------------------------------------------------------------------------
// Rank-1 state updates (float4 RMW):
//   K += p (x) o ;  KW0 += p (x) (o@Wk0) ;  KW1 += p (x) (o@Wk1)
// One block (128 thr) per (b,l) row; 4 floats per thread.
// ---------------------------------------------------------------------------
__global__ __launch_bounds__(128)
void rank1_update(float* __restrict__ K, float* __restrict__ KW0, float* __restrict__ KW1,
                  const float* __restrict__ p,
                  const float* __restrict__ o, long o_base, long o_stride,
                  const float* __restrict__ ow0, const float* __restrict__ ow1)
{
    int pair = blockIdx.x;
    int b = pair / Ll;
    float pv = p[pair];
    long rb = (long)pair * Hd;
    int h = threadIdx.x * 4;

    float4 ov = *(const float4*)(o + o_base + (long)b * o_stride + h);
    float4 w0 = *(const float4*)(ow0 + (long)b * Hd + h);
    float4 w1 = *(const float4*)(ow1 + (long)b * Hd + h);

    float4 kv = *(float4*)&K[rb + h];
    kv.x += pv * ov.x; kv.y += pv * ov.y; kv.z += pv * ov.z; kv.w += pv * ov.w;
    *(float4*)&K[rb + h] = kv;

    float4 a0 = *(float4*)&KW0[rb + h];
    a0.x += pv * w0.x; a0.y += pv * w0.y; a0.z += pv * w0.z; a0.w += pv * w0.w;
    *(float4*)&KW0[rb + h] = a0;

    float4 a1 = *(float4*)&KW1[rb + h];
    a1.x += pv * w1.x; a1.y += pv * w1.y; a1.z += pv * w1.z; a1.w += pv * w1.w;
    *(float4*)&KW1[rb + h] = a1;
}

// ---------------------------------------------------------------------------
// GRU combine (torch gate order r,z,n); biases folded into gi/gh GEMMs.
// ---------------------------------------------------------------------------
__global__ __launch_bounds__(256)
void gru_combine(const float* __restrict__ gi, const float* __restrict__ gh,
                 float* __restrict__ hbuf, float* __restrict__ out_in, int t)
{
    int i = blockIdx.x * 256 + threadIdx.x;      // over B*H
    int b = i >> 9, h = i & 511;
    long g = (long)b * 3 * Hd + h;
    float r = sigmoidf_(gi[g] + gh[g]);
    float z = sigmoidf_(gi[g + Hd] + gh[g + Hd]);
    float n = tanhf(gi[g + 2 * Hd] + r * gh[g + 2 * Hd]);
    float hn = (1.0f - z) * n + z * hbuf[i];
    hbuf[i] = hn;
    out_in[((long)b * Tt + t) * OUTIN + RNNIN + h] = hn;
}

// ---------------------------------------------------------------------------
// Gates: p_gen over out_in row; p_con over [kb_o, h]; pt_con over kbt_o.
// One block per (b,t) row; float4 streams.
// ---------------------------------------------------------------------------
__global__ __launch_bounds__(256)
void gates_kernel(const float* __restrict__ out_in,
                  const float* __restrict__ gate_w, const float* __restrict__ gate_b,
                  const float* __restrict__ copy_w, const float* __restrict__ copy_b,
                  const float* __restrict__ ckbt_w, const float* __restrict__ ckbt_b,
                  float* __restrict__ pgen, float* __restrict__ pcon, float* __restrict__ ptcon)
{
    __shared__ float r1[256], r2[256], r3[256];
    int row = blockIdx.x, tid = threadIdx.x;
    const float4* x4 = (const float4*)(out_in + (long)row * OUTIN);
    const float4* gw4 = (const float4*)gate_w;
    const float4* cw4 = (const float4*)copy_w;
    const float4* tw4 = (const float4*)ckbt_w;
    float s1 = 0.f, s2 = 0.f, s3 = 0.f;
    for (int j = tid; j < OUTIN / 4; j += 256) {
        float4 a = x4[j], w = gw4[j];
        s1 += a.x * w.x + a.y * w.y + a.z * w.z + a.w * w.w;
    }
    for (int j = tid; j < Hd / 4; j += 256) {
        float4 kb = x4[256 + j], hh = x4[512 + j], kt = x4[384 + j];
        float4 c0 = cw4[j], c1 = cw4[128 + j], tw = tw4[j];
        s2 += kb.x * c0.x + kb.y * c0.y + kb.z * c0.z + kb.w * c0.w
            + hh.x * c1.x + hh.y * c1.y + hh.z * c1.z + hh.w * c1.w;
        s3 += kt.x * tw.x + kt.y * tw.y + kt.z * tw.z + kt.w * tw.w;
    }
    r1[tid] = s1; r2[tid] = s2; r3[tid] = s3; __syncthreads();
    for (int s = 128; s > 0; s >>= 1) {
        if (tid < s) { r1[tid] += r1[tid + s]; r2[tid] += r2[tid + s]; r3[tid] += r3[tid + s]; }
        __syncthreads();
    }
    if (tid == 0) {
        pgen[row]  = sigmoidf_(r1[0] + gate_b[0]);
        pcon[row]  = sigmoidf_(r2[0] + copy_b[0]);
        ptcon[row] = sigmoidf_(r3[0] + ckbt_b[0]);
    }
}

// ---------------------------------------------------------------------------
// In-place rowwise softmax over N (vocab), float4 streams. One block per row.
// ---------------------------------------------------------------------------
__global__ __launch_bounds__(256)
void softmax_rows(float* __restrict__ data, int N)
{
    __shared__ float red[256];
    float4* row4 = (float4*)(data + (long)blockIdx.x * N);
    int tid = threadIdx.x, N4 = N / 4;
    float mx = -1e30f;
    for (int j = tid; j < N4; j += 256) {
        float4 v = row4[j];
        mx = fmaxf(mx, fmaxf(fmaxf(v.x, v.y), fmaxf(v.z, v.w)));
    }
    red[tid] = mx; __syncthreads();
    for (int s = 128; s > 0; s >>= 1) {
        if (tid < s) red[tid] = fmaxf(red[tid], red[tid + s]);
        __syncthreads();
    }
    mx = red[0]; __syncthreads();
    float sum = 0.0f;
    for (int j = tid; j < N4; j += 256) {
        float4 v = row4[j];
        sum += __expf(v.x - mx) + __expf(v.y - mx) + __expf(v.z - mx) + __expf(v.w - mx);
    }
    red[tid] = sum; __syncthreads();
    for (int s = 128; s > 0; s >>= 1) {
        if (tid < s) red[tid] += red[tid + s];
        __syncthreads();
    }
    float inv = 1.0f / red[0];
    for (int j = tid; j < N4; j += 256) {
        float4 v = row4[j];
        v.x = __expf(v.x - mx) * inv; v.y = __expf(v.y - mx) * inv;
        v.z = __expf(v.z - mx) * inv; v.w = __expf(v.w - mx) * inv;
        row4[j] = v;
    }
}

// ===========================================================================
extern "C" void kernel_launch(void* const* d_in, const int* in_sizes, int n_in,
                              void* d_out, int out_size, void* d_ws, size_t ws_size,
                              hipStream_t stream)
{
    // --- input pointers (setup_inputs dict order, params dict flattened) ---
    const int*   tokens    = (const int*)  d_in[0];
    const float* hidden0   = (const float*)d_in[1];
    const float* dlgK0     = (const float*)d_in[2];
    const float* dlgHist   = (const float*)d_in[3];
    const unsigned char* dlgMask = (const unsigned char*)d_in[4];
    const float* kbK0      = (const float*)d_in[5];
    const float* kbHist    = (const float*)d_in[6];
    const unsigned char* kbMask  = (const unsigned char*)d_in[7];
    const float* kbtMem    = (const float*)d_in[8];
    const unsigned char* kbtMask = (const unsigned char*)d_in[9];
    const float* embed     = (const float*)d_in[10];
    const float* dlg_Wq    = (const float*)d_in[11];
    const float* dlg_Wk    = (const float*)d_in[12];
    const float* dlg_b     = (const float*)d_in[13];
    const float* dlg_v     = (const float*)d_in[14];
    const float* kb_Wq     = (const float*)d_in[15];
    const float* kb_Wk     = (const float*)d_in[16];
    const float* kb_b      = (const float*)d_in[17];
    const float* kb_v      = (const float*)d_in[18];
    const float* kbt_Wq    = (const float*)d_in[19];
    const float* kbt_Wk    = (const float*)d_in[20];
    const float* kbt_b     = (const float*)d_in[21];
    const float* kbt_v     = (const float*)d_in[22];
    const float* W_ih      = (const float*)d_in[23];
    const float* W_hh      = (const float*)d_in[24];
    const float* b_ih      = (const float*)d_in[25];
    const float* b_hh      = (const float*)d_in[26];
    const float* gate_w    = (const float*)d_in[27];
    const float* gate_b    = (const float*)d_in[28];
    const float* copy_w    = (const float*)d_in[29];
    const float* copy_b    = (const float*)d_in[30];
    const float* ckbt_w    = (const float*)d_in[31];
    const float* ckbt_b    = (const float*)d_in[32];
    const float* out_w1    = (const float*)d_in[33];
    const float* out_b1    = (const float*)d_in[34];
    const float* out_w2    = (const float*)d_in[35];
    const float* out_b2    = (const float*)d_in[36];

    float* out = (float*)d_out;
    // --- d_out layout (floats) ---
    const long O_PROBS = 0;
    const long O_PGEN  = (long)Bb * Tt * Vv;            // 65,536,000
    const long O_PCON  = O_PGEN + (long)Bb * Tt;
    const long O_PTCON = O_PCON + (long)Bb * Tt;
    const long O_DLGP  = O_PTCON + (long)Bb * Tt;
    const long O_KBP   = O_DLGP + (long)Bb * Tt * Ll;
    const long O_KBTP  = O_KBP + (long)Bb * Tt * Ll;
    const long O_HF    = O_KBTP + (long)Bb * Tt * Ll;
    const long O_DK    = O_HF + (long)Bb * Hd;
    const long O_KK    = O_DK + (long)Bb * Ll * Hd;

    float* hbuf = out + O_HF;   // mutable hidden lives in its output slot
    float* dlgK = out + O_DK;   // mutable dlg state memory (final = dK_f)
    float* kbK  = out + O_KK;   // mutable kb state memory (final = kK_f)

    // --- workspace layout (floats) ---
    float* ws = (float*)d_ws;
    long off = 0;
    float* out_in = ws + off; off += (long)Bb * Tt * OUTIN;   // 5,242,880
    float* KWD0   = ws + off; off += (long)Bb * Ll * Hd;
    float* KWD1   = ws + off; off += (long)Bb * Ll * Hd;
    float* KWK0   = ws + off; off += (long)Bb * Ll * Hd;
    float* KWK1   = ws + off; off += (long)Bb * Ll * Hd;
    float* KWT    = ws + off; off += (long)Bb * Ll * Hd;
    float* qtmp   = ws + off; off += (long)Bb * Hd;
    float* qW     = ws + off; off += (long)Bb * Hd;
    float* ebuf   = ws + off; off += (long)Bb * Ll;
    float* pbuf   = ws + off; off += (long)Bb * Ll;
    float* owb0   = ws + off; off += (long)Bb * Hd;
    float* owb1   = ws + off; off += (long)Bb * Hd;
    float* gi     = ws + off; off += (long)Bb * 3 * Hd;
    float* gh     = ws + off; off += (long)Bb * 3 * Hd;
    float* hid1   = ws + off; off += (long)Bb * Tt * Hd;      // 1,048,576

    const long LDA_STEP = (long)Tt * OUTIN;   // batch-row stride inside out_in slices
    const long HH = (long)Hd * Hd;

    auto gemm = [&](const float* A, long abase, int lda,
                    const float* Bw, int ldb, int transB, const float* bias,
                    float* C, long cbase, int ldc, int M, int N, int K) {
        dim3 g((N + TILE_N - 1) / TILE_N, (M + TILE_M - 1) / TILE_M);
        gemm_wmma_bf16<<<g, dim3(128), 0, stream>>>(A, abase, lda, Bw, ldb, transB,
                                                    bias, C, cbase, ldc, M, N, K);
    };

    // === init mutable state in d_out ===
    hipMemcpyAsync(hbuf, hidden0, (size_t)Bb * Hd * sizeof(float), hipMemcpyDeviceToDevice, stream);
    hipMemcpyAsync(dlgK, dlgK0, (size_t)Bb * Ll * Hd * sizeof(float), hipMemcpyDeviceToDevice, stream);
    hipMemcpyAsync(kbK,  kbK0,  (size_t)Bb * Ll * Hd * sizeof(float), hipMemcpyDeviceToDevice, stream);

    // === embeddings for all (b,t) straight into out_in[:, :, 0:512] ===
    embed_gather<<<dim3((Bb * Tt * Ed / 4) / 256), dim3(256), 0, stream>>>(tokens, embed, out_in);

    // === hoisted K@Wk + b precompute (WMMA), rank-1-updated thereafter ===
    gemm(dlgK0, 0, Hd, dlg_Wk,        Hd, 0, dlg_b,      KWD0, 0, Hd, Bb * Ll, Hd, Hd);
    gemm(dlgK0, 0, Hd, dlg_Wk + HH,   Hd, 0, dlg_b + Hd, KWD1, 0, Hd, Bb * Ll, Hd, Hd);
    gemm(kbK0,  0, Hd, kb_Wk,         Hd, 0, kb_b,       KWK0, 0, Hd, Bb * Ll, Hd, Hd);
    gemm(kbK0,  0, Hd, kb_Wk + HH,    Hd, 0, kb_b + Hd,  KWK1, 0, Hd, Bb * Ll, Hd, Hd);
    gemm(kbtMem,0, Hd, kbt_Wk,        Hd, 0, kbt_b,      KWT,  0, Hd, Bb * Ll, Hd, Hd);

    // === sequential decode loop ===
    for (int t = 0; t < Tt; ++t) {
        const long ob_dlg = (long)t * OUTIN + 512;
        const long ob_kb  = (long)t * OUTIN + 1024;
        const long ob_kbt = (long)t * OUTIN + 1536;

        // ---- dialogue memory: 2-hop MLP attention + rank-1 write ----
        gemm(hbuf, 0, Hd, dlg_Wq, Hd, 0, nullptr, qW, 0, Hd, Bb, Hd, Hd);
        attn_scores<<<dim3(Bb * Ll / 8), dim3(256), 0, stream>>>(qW, KWD0, dlg_v, dlgMask, ebuf);
        softmax_ctx<<<dim3(Bb), dim3(256), 0, stream>>>(ebuf, dlgHist, pbuf,
            nullptr, 0, hbuf, qtmp, nullptr, 0, 0);
        gemm(qtmp, 0, Hd, dlg_Wq + HH, Hd, 0, nullptr, qW, 0, Hd, Bb, Hd, Hd);
        attn_scores<<<dim3(Bb * Ll / 8), dim3(256), 0, stream>>>(qW, KWD1, dlg_v + Hd, dlgMask, ebuf);
        softmax_ctx<<<dim3(Bb), dim3(256), 0, stream>>>(ebuf, dlgHist, pbuf,
            out + O_DLGP + (long)t * Ll, (long)Tt * Ll,
            nullptr, nullptr, out_in, ob_dlg, LDA_STEP);
        gemm(out_in, ob_dlg, (int)LDA_STEP, dlg_Wk,      Hd, 0, nullptr, owb0, 0, Hd, Bb, Hd, Hd);
        gemm(out_in, ob_dlg, (int)LDA_STEP, dlg_Wk + HH, Hd, 0, nullptr, owb1, 0, Hd, Bb, Hd, Hd);
        rank1_update<<<dim3(Bb * Ll), dim3(128), 0, stream>>>(dlgK, KWD0, KWD1, pbuf,
            out_in, ob_dlg, LDA_STEP, owb0, owb1);

        // ---- KB memory: 2-hop MLP attention + rank-1 write ----
        gemm(hbuf, 0, Hd, kb_Wq, Hd, 0, nullptr, qW, 0, Hd, Bb, Hd, Hd);
        attn_scores<<<dim3(Bb * Ll / 8), dim3(256), 0, stream>>>(qW, KWK0, kb_v, kbMask, ebuf);
        softmax_ctx<<<dim3(Bb), dim3(256), 0, stream>>>(ebuf, kbHist, pbuf,
            nullptr, 0, hbuf, qtmp, nullptr, 0, 0);
        gemm(qtmp, 0, Hd, kb_Wq + HH, Hd, 0, nullptr, qW, 0, Hd, Bb, Hd, Hd);
        attn_scores<<<dim3(Bb * Ll / 8), dim3(256), 0, stream>>>(qW, KWK1, kb_v + Hd, kbMask, ebuf);
        softmax_ctx<<<dim3(Bb), dim3(256), 0, stream>>>(ebuf, kbHist, pbuf,
            out + O_KBP + (long)t * Ll, (long)Tt * Ll,
            nullptr, nullptr, out_in, ob_kb, LDA_STEP);
        gemm(out_in, ob_kb, (int)LDA_STEP, kb_Wk,      Hd, 0, nullptr, owb0, 0, Hd, Bb, Hd, Hd);
        gemm(out_in, ob_kb, (int)LDA_STEP, kb_Wk + HH, Hd, 0, nullptr, owb1, 0, Hd, Bb, Hd, Hd);
        rank1_update<<<dim3(Bb * Ll), dim3(128), 0, stream>>>(kbK, KWK0, KWK1, pbuf,
            out_in, ob_kb, LDA_STEP, owb0, owb1);

        // ---- kbt memory: single-hop read (static K@Wk) ----
        gemm(hbuf, 0, Hd, kbt_Wq, Hd, 0, nullptr, qW, 0, Hd, Bb, Hd, Hd);
        attn_scores<<<dim3(Bb * Ll / 8), dim3(256), 0, stream>>>(qW, KWT, kbt_v, kbtMask, ebuf);
        softmax_ctx<<<dim3(Bb), dim3(256), 0, stream>>>(ebuf, kbtMem, pbuf,
            out + O_KBTP + (long)t * Ll, (long)Tt * Ll,
            nullptr, nullptr, out_in, ob_kbt, LDA_STEP);

        // ---- GRU cell: gi = rnn_in @ W_ih^T + b_ih ; gh = h @ W_hh^T + b_hh ----
        gemm(out_in, (long)t * OUTIN, (int)LDA_STEP, W_ih, RNNIN, 1, b_ih,
             gi, 0, 3 * Hd, Bb, 3 * Hd, RNNIN);
        gemm(hbuf, 0, Hd, W_hh, Hd, 1, b_hh, gh, 0, 3 * Hd, Bb, 3 * Hd, Hd);
        gru_combine<<<dim3(Bb * Hd / 256), dim3(256), 0, stream>>>(gi, gh, hbuf, out_in, t);
    }

    // === output head: (out_in @ W1 + b1) @ W2 + b2 -> softmax (in d_out) ===
    gemm(out_in, 0, OUTIN, out_w1, Hd, 0, out_b1, hid1, 0, Hd, Bb * Tt, Hd, OUTIN);
    gemm(hid1, 0, Hd, out_w2, Vv, 0, out_b2, out + O_PROBS, 0, Vv, Bb * Tt, Vv, Hd);
    softmax_rows<<<dim3(Bb * Tt), dim3(256), 0, stream>>>(out + O_PROBS, Vv);

    // === pointer-mix gates ===
    gates_kernel<<<dim3(Bb * Tt), dim3(256), 0, stream>>>(out_in, gate_w, gate_b,
        copy_w, copy_b, ckbt_w, ckbt_b, out + O_PGEN, out + O_PCON, out + O_PTCON);
}
